// EGCAN_32392643346785
// MI455X (gfx1250) — compile-verified, matrix operand order checked by reading, not run on hardware
//
#include <hip/hip_runtime.h>
#include <hip/hip_bf16.h>
#include <math.h>

#define N_NODES 50000
#define N_EDGES 800000
#define ETOT    (N_EDGES + N_NODES)
#define IN_C    512
#define EMB     1000
#define HID     64
#define OUT_C   20

#define USE_ASYNC 1

typedef __attribute__((ext_vector_type(16))) __bf16 v16b;
typedef __attribute__((ext_vector_type(8)))  __bf16 v8b;
typedef __attribute__((ext_vector_type(8)))  float  v8f;
typedef __attribute__((ext_vector_type(4)))  unsigned v4u;

// fp32 -> bf16 bits, round-to-nearest-even
__device__ __forceinline__ unsigned short bfbits(float f) {
  unsigned u = __float_as_uint(f);
  return (unsigned short)((u + 0x7FFFu + ((u >> 16) & 1u)) >> 16);
}
__device__ __forceinline__ unsigned pack2bf(float a, float b) {
  unsigned ua = __float_as_uint(a);
  unsigned ub = __float_as_uint(b);
  ua = (ua + 0x7FFFu + ((ua >> 16) & 1u)) >> 16;
  ub = (ub + 0x7FFFu + ((ub >> 16) & 1u)) & 0xFFFF0000u;
  return ua | ub;
}

#if USE_ASYNC
// 16B/lane global -> LDS async copy (ASYNCcnt); LDS byte addr = low 32 bits of flat addr
__device__ __forceinline__ void async_copy_b128(unsigned lds_addr, const void* gaddr) {
  asm volatile("global_load_async_to_lds_b128 %0, %1, off"
               :: "v"(lds_addr), "v"((unsigned long long)(size_t)gaddr)
               : "memory");
}
#define S_WAIT_ASYNC(N) asm volatile("s_wait_asynccnt " #N ::: "memory")
#endif

// ------- WMMA bf16 GEMM: C[M,Nc] = act(A[M,K] @ Bt[Nc,K]^T + bias) ---------
// A: bf16 row-major [M,K]; Bt: bf16 row-major [Nc,K] (pre-transposed B).
// Requires K % 8 == 0 (all call sites: 512/1000/64).
#define TM 64
#define TN 128
#define TK 32
#define LDK 40   // padded K-stride (80B rows -> 16B aligned, conflict-free)

__global__ __launch_bounds__(256) void wmma_gemm_bf16(
    const unsigned short* __restrict__ A, const unsigned short* __restrict__ Bt,
    const float* __restrict__ bias, void* __restrict__ Cout,
    int M, int K, int Nc, int act, int outBf16)
{
  __shared__ __bf16 As[2][TM * LDK];
  __shared__ __bf16 Bs[2][TN * LDK];
  const int tid  = threadIdx.x;
  const int lane = tid & 31;
  const int wave = tid >> 5;
  const int wr   = wave >> 2;          // 0..1 (32-row group)
  const int wc   = wave & 3;           // 0..3 (32-col group)
  const int l16  = lane & 15;
  const int lhi  = lane >> 4;
  const int koff = lhi << 3;

  const int mBase = blockIdx.y * TM;
  const int nBase = blockIdx.x * TN;

  // chunk mapping: 16B (8 bf16) per lane-chunk
  const int cr  = tid >> 2;            // 0..63  (A row / B row pair base)
  const int cks = (tid & 3) * 8;       // k offset within tile
  // clamped source rows: OOB rows read valid duplicate data; results discarded
  const int amc = (mBase + cr       < M)  ? (mBase + cr)       : (M - 1);
  const int bn0 = (nBase + cr       < Nc) ? (nBase + cr)       : (Nc - 1);
  const int bn1 = (nBase + cr + 64  < Nc) ? (nBase + cr + 64)  : (Nc - 1);

  auto stageFast = [&](int k0, int buf) {   // full K-chunk stage
#if USE_ASYNC
    async_copy_b128((unsigned)(size_t)&As[buf][cr * LDK + cks],
                    A  + (size_t)amc * K + k0 + cks);
    async_copy_b128((unsigned)(size_t)&Bs[buf][cr * LDK + cks],
                    Bt + (size_t)bn0 * K + k0 + cks);
    async_copy_b128((unsigned)(size_t)&Bs[buf][(cr + 64) * LDK + cks],
                    Bt + (size_t)bn1 * K + k0 + cks);
#else
    *(v8b*)&As[buf][cr * LDK + cks]        = *(const v8b*)(A  + (size_t)amc * K + k0 + cks);
    *(v8b*)&Bs[buf][cr * LDK + cks]        = *(const v8b*)(Bt + (size_t)bn0 * K + k0 + cks);
    *(v8b*)&Bs[buf][(cr + 64) * LDK + cks] = *(const v8b*)(Bt + (size_t)bn1 * K + k0 + cks);
#endif
  };

  auto stageSlow = [&](int k0, int buf) {   // ragged K tail: zero-pad, sync stores
    unsigned short e[8];
    v4u pk;
    #pragma unroll
    for (int i = 0; i < 8; ++i) { int k = k0 + cks + i; e[i] = (k < K) ? A[(size_t)amc * K + k] : 0; }
    pk[0] = e[0] | (e[1] << 16); pk[1] = e[2] | (e[3] << 16);
    pk[2] = e[4] | (e[5] << 16); pk[3] = e[6] | (e[7] << 16);
    *(v4u*)&As[buf][cr * LDK + cks] = pk;
    #pragma unroll
    for (int i = 0; i < 8; ++i) { int k = k0 + cks + i; e[i] = (k < K) ? Bt[(size_t)bn0 * K + k] : 0; }
    pk[0] = e[0] | (e[1] << 16); pk[1] = e[2] | (e[3] << 16);
    pk[2] = e[4] | (e[5] << 16); pk[3] = e[6] | (e[7] << 16);
    *(v4u*)&Bs[buf][cr * LDK + cks] = pk;
    #pragma unroll
    for (int i = 0; i < 8; ++i) { int k = k0 + cks + i; e[i] = (k < K) ? Bt[(size_t)bn1 * K + k] : 0; }
    pk[0] = e[0] | (e[1] << 16); pk[1] = e[2] | (e[3] << 16);
    pk[2] = e[4] | (e[5] << 16); pk[3] = e[6] | (e[7] << 16);
    *(v4u*)&Bs[buf][(cr + 64) * LDK + cks] = pk;
  };

  v8f acc00 = {}; v8f acc01 = {}; v8f acc10 = {}; v8f acc11 = {};

  const int nStages = (K + TK - 1) / TK;
  stageFast(0, 0);                      // K >= TK at all call sites

  for (int s = 0; s < nStages; ++s) {
    const int cur = s & 1;
    bool issuedAsync = false;
    if (s + 1 < nStages) {
      int k0n = (s + 1) * TK;
      if (k0n + TK <= K) { stageFast(k0n, cur ^ 1); issuedAsync = true; }
      else              {  stageSlow(k0n, cur ^ 1); }
    }
#if USE_ASYNC
    if (issuedAsync) { S_WAIT_ASYNC(3); } else { S_WAIT_ASYNC(0); }
#else
    (void)issuedAsync;
#endif
    __syncthreads();                    // stage s data visible to all waves

    v16b a0, a1, b0, b1;
    {
      const __bf16* p;
      v8b lo, hi;
      p = &As[cur][(wr * 32 + l16) * LDK];
      lo = *(const v8b*)(p + koff); hi = *(const v8b*)(p + 16 + koff);
      a0 = __builtin_shufflevector(lo, hi, 0,1,2,3,4,5,6,7,8,9,10,11,12,13,14,15);
      p = &As[cur][(wr * 32 + 16 + l16) * LDK];
      lo = *(const v8b*)(p + koff); hi = *(const v8b*)(p + 16 + koff);
      a1 = __builtin_shufflevector(lo, hi, 0,1,2,3,4,5,6,7,8,9,10,11,12,13,14,15);
      p = &Bs[cur][(wc * 32 + l16) * LDK];
      lo = *(const v8b*)(p + koff); hi = *(const v8b*)(p + 16 + koff);
      b0 = __builtin_shufflevector(lo, hi, 0,1,2,3,4,5,6,7,8,9,10,11,12,13,14,15);
      p = &Bs[cur][(wc * 32 + 16 + l16) * LDK];
      lo = *(const v8b*)(p + koff); hi = *(const v8b*)(p + 16 + koff);
      b1 = __builtin_shufflevector(lo, hi, 0,1,2,3,4,5,6,7,8,9,10,11,12,13,14,15);
    }
    acc00 = __builtin_amdgcn_wmma_f32_16x16x32_bf16(false, a0, false, b0, (short)0, acc00, false, false);
    acc01 = __builtin_amdgcn_wmma_f32_16x16x32_bf16(false, a0, false, b1, (short)0, acc01, false, false);
    acc10 = __builtin_amdgcn_wmma_f32_16x16x32_bf16(false, a1, false, b0, (short)0, acc10, false, false);
    acc11 = __builtin_amdgcn_wmma_f32_16x16x32_bf16(false, a1, false, b1, (short)0, acc11, false, false);
    __syncthreads();                    // readers done before next stage's writes land
  }

  const float SC = 1.0507009873554805f, AL = 1.6732632423543772f;
  #pragma unroll
  for (int i = 0; i < 2; ++i) {
    #pragma unroll
    for (int j = 0; j < 2; ++j) {
      v8f acc = (i == 0) ? (j == 0 ? acc00 : acc01) : (j == 0 ? acc10 : acc11);
      int n = nBase + wc * 32 + j * 16 + l16;
      #pragma unroll
      for (int r = 0; r < 8; ++r) {
        int m = mBase + wr * 32 + i * 16 + r + (lhi << 3);
        if (m < M && n < Nc) {
          float v = acc[r];
          if (bias) v += bias[n];
          if (act == 1) v = (v > 0.0f) ? SC * v : SC * AL * (expf(v) - 1.0f);
          if (outBf16) ((unsigned short*)Cout)[(size_t)m * Nc + n] = bfbits(v);
          else         ((float*)Cout)[(size_t)m * Nc + n] = v;
        }
      }
    }
  }
}

// ---------------- conversion / transpose kernels ---------------------------
__global__ void k_cvt_pair(const float* __restrict__ in, unsigned* __restrict__ outp, int npairs) {
  int i = blockIdx.x * blockDim.x + threadIdx.x;
  if (i < npairs) outp[i] = pack2bf(in[2 * i], in[2 * i + 1]);
}
// W[K,Nc] fp32 -> Wt[Nc,K] bf16
__global__ void k_transpose_bf16(const float* __restrict__ W, unsigned short* __restrict__ Wt,
                                 int K, int Nc) {
  int idx = blockIdx.x * blockDim.x + threadIdx.x;
  if (idx < K * Nc) { int k = idx / Nc; int n = idx - k * Nc;
    Wt[(size_t)n * K + k] = bfbits(W[idx]); }
}

// ---------------- utility / graph kernels ----------------------------------
__global__ void k_fill_f32(float* p, float v, int n) {
  int i = blockIdx.x * blockDim.x + threadIdx.x; if (i < n) p[i] = v;
}
__global__ void k_fill_u32(unsigned* p, unsigned v, int n) {
  int i = blockIdx.x * blockDim.x + threadIdx.x; if (i < n) p[i] = v;
}
__global__ void k_deg(const int* __restrict__ dst, const float* __restrict__ ew,
                      float* deg, int E) {
  int e = blockIdx.x * blockDim.x + threadIdx.x;
  if (e < E) atomicAdd(&deg[dst[e]], ew[e]);
}
__global__ void k_dis(const float* __restrict__ deg, float* dis, int n) {
  int i = blockIdx.x * blockDim.x + threadIdx.x;
  if (i < n) { float d = deg[i] + 1.0f;
    dis[i] = d > 0.0f ? rsqrtf(fmaxf(d, 1e-12f)) : 0.0f; }
}
__global__ void k_gcn_init(float* out, const float* __restrict__ t,
                           const float* __restrict__ bias, const float* __restrict__ dis, int n) {
  int i = blockIdx.x * blockDim.x + threadIdx.x;
  if (i < n) { int node = i >> 6; int f = i & 63; float s = dis[node];
    out[i] = bias[f] + t[i] * s * s; }
}
__global__ void k_gcn_edge(float* out, const float* __restrict__ t, const float* __restrict__ dis,
                           const int* __restrict__ src, const int* __restrict__ dst,
                           const float* __restrict__ ew, int E) {
  int idx = blockIdx.x * blockDim.x + threadIdx.x;
  if (idx < E * 64) {
    int e = idx >> 6, f = idx & 63;
    int s = src[e], d = dst[e];
    float nrm = dis[s] * ew[e] * dis[d];
    atomicAdd(&out[d * 64 + f], t[s * 64 + f] * nrm);
  }
}
__global__ void k_gat_al(const float* __restrict__ hh, const float* __restrict__ asrc,
                         const float* __restrict__ adst, float* als, float* ald, int n) {
  int i = blockIdx.x * blockDim.x + threadIdx.x;
  if (i < n) { int node = i >> 1, h = i & 1;
    const float* p = hh + (size_t)node * 40 + h * 20;
    float s = 0.f, d = 0.f;
    #pragma unroll
    for (int c = 0; c < 20; ++c) { float v = p[c]; s += v * asrc[h * 20 + c]; d += v * adst[h * 20 + c]; }
    als[i] = s; ald[i] = d; }
}
__device__ __forceinline__ unsigned fenc(float f) {
  unsigned u = __float_as_uint(f);
  return (u & 0x80000000u) ? ~u : (u | 0x80000000u);
}
__device__ __forceinline__ float fdec(unsigned u) {
  unsigned b = (u & 0x80000000u) ? (u ^ 0x80000000u) : ~u;
  return __uint_as_float(b);
}
__global__ void k_alpha_max(const int* __restrict__ src, const int* __restrict__ dst,
                            const float* __restrict__ als, const float* __restrict__ ald,
                            float* alpha, unsigned* mEnc, int n) {
  int i = blockIdx.x * blockDim.x + threadIdx.x;
  if (i < n) { int e = i >> 1, h = i & 1;
    int s, d;
    if (e < N_EDGES) { s = src[e]; d = dst[e]; } else { s = d = e - N_EDGES; }
    float a = als[s * 2 + h] + ald[d * 2 + h];
    a = a > 0.0f ? a : 0.2f * a;
    alpha[i] = a;
    atomicMax(&mEnc[d * 2 + h], fenc(a)); }
}
__global__ void k_den(const int* __restrict__ dst, const unsigned* __restrict__ mEnc,
                      float* alpha, float* den, int n) {
  int i = blockIdx.x * blockDim.x + threadIdx.x;
  if (i < n) { int e = i >> 1, h = i & 1;
    int d = (e < N_EDGES) ? dst[e] : e - N_EDGES;
    float ex = expf(alpha[i] - fdec(mEnc[d * 2 + h]));
    alpha[i] = ex;
    atomicAdd(&den[d * 2 + h], ex); }
}
__global__ void k_msg(const int* __restrict__ src, const int* __restrict__ dst,
                      const float* __restrict__ hh, const float* __restrict__ alpha,
                      const float* __restrict__ den, float* agg, int total) {
  int i = blockIdx.x * blockDim.x + threadIdx.x;
  if (i < total) {
    int e = i / 40; int r = i - e * 40; int h = r / 20; int c = r - h * 20;
    int s, d;
    if (e < N_EDGES) { s = src[e]; d = dst[e]; } else { s = d = e - N_EDGES; }
    float att = alpha[e * 2 + h] / (den[d * 2 + h] + 1e-16f);
    atomicAdd(&agg[(size_t)d * 40 + h * 20 + c], hh[(size_t)s * 40 + h * 20 + c] * att);
  }
}
__global__ void k_final(const float* __restrict__ agg, const float* __restrict__ bg,
                        float* out, int n) {
  int i = blockIdx.x * blockDim.x + threadIdx.x;
  if (i < n) { int node = i / 20; int c = i - node * 20;
    out[i] = 0.5f * (agg[(size_t)node * 40 + c] + agg[(size_t)node * 40 + 20 + c]) + bg[c]; }
}

// ---------------- host launch ----------------------------------------------
extern "C" void kernel_launch(void* const* d_in, const int* in_sizes, int n_in,
                              void* d_out, int out_size, void* d_ws, size_t ws_size,
                              hipStream_t stream) {
  const float* x    = (const float*)d_in[0];
  const int*   ei   = (const int*)  d_in[1];
  const float* ew   = (const float*)d_in[2];
  const float* W0   = (const float*)d_in[3];
  const float* b0   = (const float*)d_in[4];
  const float* W1   = (const float*)d_in[5];
  const float* b1   = (const float*)d_in[6];
  const float* W2   = (const float*)d_in[7];
  const float* b2   = (const float*)d_in[8];
  const float* Wg   = (const float*)d_in[9];
  const float* asrc = (const float*)d_in[10];
  const float* adst = (const float*)d_in[11];
  const float* bg   = (const float*)d_in[12];
  const int* src = ei;
  const int* dst = ei + N_EDGES;
  float* out = (float*)d_out;
  (void)in_sizes; (void)n_in; (void)out_size; (void)ws_size;

  char* base = (char*)d_ws;
  size_t off = 0;
  auto allocB = [&](size_t bytes) -> void* {
    void* p = (void*)(base + off);
    off += (bytes + 255) & ~(size_t)255;
    return p;
  };
  unsigned short* xb   = (unsigned short*)allocB((size_t)N_NODES * IN_C * 2);
  unsigned short* h1b  = (unsigned short*)allocB((size_t)N_NODES * EMB * 2);
  unsigned short* W0t  = (unsigned short*)allocB((size_t)EMB * IN_C * 2);
  unsigned short* W1t  = (unsigned short*)allocB((size_t)HID * EMB * 2);
  unsigned short* W2t  = (unsigned short*)allocB((size_t)HID * HID * 2);
  unsigned short* Wgt  = (unsigned short*)allocB((size_t)40 * HID * 2);
  unsigned short* h2b  = (unsigned short*)allocB((size_t)N_NODES * HID * 2);
  unsigned short* h3b  = (unsigned short*)allocB((size_t)N_NODES * HID * 2);
  float*    tb    = (float*)allocB((size_t)N_NODES * HID * 4);
  float*    h2    = (float*)allocB((size_t)N_NODES * HID * 4);
  float*    h3    = (float*)allocB((size_t)N_NODES * HID * 4);
  float*    deg   = (float*)allocB((size_t)N_NODES * 4);
  float*    dis   = (float*)allocB((size_t)N_NODES * 4);
  float*    hh    = (float*)allocB((size_t)N_NODES * 40 * 4);
  float*    als   = (float*)allocB((size_t)N_NODES * 2 * 4);
  float*    ald   = (float*)allocB((size_t)N_NODES * 2 * 4);
  unsigned* mEnc  = (unsigned*)allocB((size_t)N_NODES * 2 * 4);
  float*    den   = (float*)allocB((size_t)N_NODES * 2 * 4);
  float*    alpha = (float*)allocB((size_t)ETOT * 2 * 4);
  float*    agg   = (float*)allocB((size_t)N_NODES * 40 * 4);

  const int T = 256;
  auto nb = [&](long long n) { return (unsigned)((n + T - 1) / T); };

  // precision prep: bf16-resident A operands + transposed bf16 weights
  k_cvt_pair<<<nb((long long)N_NODES * IN_C / 2), T, 0, stream>>>(x, (unsigned*)xb, N_NODES * IN_C / 2);
  k_transpose_bf16<<<nb((long long)IN_C * EMB), T, 0, stream>>>(W0, W0t, IN_C, EMB);
  k_transpose_bf16<<<nb((long long)EMB * HID), T, 0, stream>>>(W1, W1t, EMB, HID);
  k_transpose_bf16<<<nb((long long)HID * HID), T, 0, stream>>>(W2, W2t, HID, HID);
  k_transpose_bf16<<<nb((long long)HID * 40), T, 0, stream>>>(Wg, Wgt, HID, 40);

  // degree / normalization (shared by both GCN layers)
  k_fill_f32<<<nb(N_NODES), T, 0, stream>>>(deg, 0.0f, N_NODES);
  k_deg<<<nb(N_EDGES), T, 0, stream>>>(dst, ew, deg, N_EDGES);
  k_dis<<<nb(N_NODES), T, 0, stream>>>(deg, dis, N_NODES);

  // h1 = selu(x @ W0 + b0) -> bf16 [N, 1000]
  dim3 g0((EMB + TN - 1) / TN, (N_NODES + TM - 1) / TM);
  wmma_gemm_bf16<<<g0, 256, 0, stream>>>(xb, W0t, b0, h1b, N_NODES, IN_C, EMB, 1, 1);

  // GCN layer 1
  dim3 g1((HID + TN - 1) / TN, (N_NODES + TM - 1) / TM);
  wmma_gemm_bf16<<<g1, 256, 0, stream>>>(h1b, W1t, nullptr, tb, N_NODES, EMB, HID, 0, 0);
  k_gcn_init<<<nb((long long)N_NODES * HID), T, 0, stream>>>(h2, tb, b1, dis, N_NODES * HID);
  k_gcn_edge<<<nb((long long)N_EDGES * 64), T, 0, stream>>>(h2, tb, dis, src, dst, ew, N_EDGES);

  // GCN layer 2
  k_cvt_pair<<<nb((long long)N_NODES * HID / 2), T, 0, stream>>>(h2, (unsigned*)h2b, N_NODES * HID / 2);
  wmma_gemm_bf16<<<g1, 256, 0, stream>>>(h2b, W2t, nullptr, tb, N_NODES, HID, HID, 0, 0);
  k_gcn_init<<<nb((long long)N_NODES * HID), T, 0, stream>>>(h3, tb, b2, dis, N_NODES * HID);
  k_gcn_edge<<<nb((long long)N_EDGES * 64), T, 0, stream>>>(h3, tb, dis, src, dst, ew, N_EDGES);

  // GAT
  k_cvt_pair<<<nb((long long)N_NODES * HID / 2), T, 0, stream>>>(h3, (unsigned*)h3b, N_NODES * HID / 2);
  dim3 g2(1, (N_NODES + TM - 1) / TM);
  wmma_gemm_bf16<<<g2, 256, 0, stream>>>(h3b, Wgt, nullptr, hh, N_NODES, HID, 40, 0, 0);
  k_gat_al<<<nb((long long)N_NODES * 2), T, 0, stream>>>(hh, asrc, adst, als, ald, N_NODES * 2);
  k_fill_u32<<<nb((long long)N_NODES * 2), T, 0, stream>>>(mEnc, 0x00800000u /*enc(-FLT_MAX)*/, N_NODES * 2);
  k_fill_f32<<<nb((long long)N_NODES * 2), T, 0, stream>>>(den, 0.0f, N_NODES * 2);
  k_fill_f32<<<nb((long long)N_NODES * 40), T, 0, stream>>>(agg, 0.0f, N_NODES * 40);
  k_alpha_max<<<nb((long long)ETOT * 2), T, 0, stream>>>(src, dst, als, ald, alpha, mEnc, ETOT * 2);
  k_den<<<nb((long long)ETOT * 2), T, 0, stream>>>(dst, mEnc, alpha, den, ETOT * 2);
  k_msg<<<nb((long long)ETOT * 40), T, 0, stream>>>(src, dst, hh, alpha, den, agg, ETOT * 40);
  k_final<<<nb((long long)N_NODES * 20), T, 0, stream>>>(agg, bg, out, N_NODES * 20);
}
